// PrunedModel_19043884990715
// MI455X (gfx1250) — matrix operand order repaired and checked by reading, not compiled, not run on hardware
//
#include <hip/hip_runtime.h>
#include <cstdint>

// ---------------------------------------------------------------------------
// Problem constants (match reference)
// ---------------------------------------------------------------------------
#define HIDDEN 4096
#define PRUNED 2048
#define INTER  5004
#define INTER_P 5120          // padded to multiple of 128 for clean tiling
#define MTOK   16384          // B*S tokens
#define EPSLN  1e-5f
#define LOG2E  1.4426950408889634f

typedef __attribute__((ext_vector_type(16))) __bf16 v16bf;
typedef __attribute__((ext_vector_type(8)))  float v8f;
typedef unsigned short ushort_t;

__device__ __forceinline__ float bf2f(ushort_t u) {
    union { uint32_t i; float f; } v; v.i = ((uint32_t)u) << 16; return v.f;
}
__device__ __forceinline__ ushort_t f2bf(float f) {
    union { float f; uint32_t i; } v; v.f = f;
    uint32_t r = v.i + 0x7FFFu + ((v.i >> 16) & 1u);   // RNE
    return (ushort_t)(r >> 16);
}

union AFrag { v16bf v; uint4 q[2]; };
union CFrag { v8f v; float f[8]; };

// --- CDNA5 async global->LDS copies (ASYNCcnt-tracked, no VGPR transit) ----
__device__ __forceinline__ uint32_t lds_off(const void* p) {
    // generic pointer to LDS: low 32 bits are the LDS byte offset
    return (uint32_t)(uintptr_t)p;
}
__device__ __forceinline__ void async_b128(uint32_t dst, const void* src) {
    asm volatile("global_load_async_to_lds_b128 %0, %1, off"
                 :: "v"(dst), "v"(src) : "memory");
}
__device__ __forceinline__ void async_b64(uint32_t dst, const void* src) {
    asm volatile("global_load_async_to_lds_b64 %0, %1, off"
                 :: "v"(dst), "v"(src) : "memory");
}
__device__ __forceinline__ void wait_async0() {
    asm volatile("s_wait_asynccnt 0x0" ::: "memory");
}

// ---------------------------------------------------------------------------
// K0: residual copy
// ---------------------------------------------------------------------------
__global__ void copy_residual_kernel(const uint4* __restrict__ src,
                                     uint4* __restrict__ dst) {
    size_t i = (size_t)blockIdx.x * blockDim.x + threadIdx.x;
    dst[i] = src[i];
}

// ---------------------------------------------------------------------------
// K1: fused LayerNorm + gather(s3) -> h [MTOK x PRUNED] bf16 (block per token)
// ---------------------------------------------------------------------------
__global__ void ln_gather_kernel(const ushort_t* __restrict__ x,
                                 const ushort_t* __restrict__ lnw,
                                 const ushort_t* __restrict__ lnb,
                                 const int* __restrict__ s3,
                                 ushort_t* __restrict__ hout) {
    __shared__ float ws_s[8], ws_q[8], stat[2];
    const int tid  = threadIdx.x;
    const int lane = tid & 31;
    const int wave = tid >> 5;
    const ushort_t* row = x + (size_t)blockIdx.x * HIDDEN;
    const uint4* rp = (const uint4*)row;      // 512 uint4 per row

    float s = 0.f, ss = 0.f;
    for (int i = tid; i < HIDDEN / 8; i += 256) {
        union { uint4 q; ushort_t h[8]; } d; d.q = rp[i];
        #pragma unroll
        for (int e = 0; e < 8; ++e) {
            float f = bf2f(d.h[e]);
            s += f; ss += f * f;
        }
    }
    #pragma unroll
    for (int off = 16; off > 0; off >>= 1) {
        s  += __shfl_xor(s,  off, 32);
        ss += __shfl_xor(ss, off, 32);
    }
    if (lane == 0) { ws_s[wave] = s; ws_q[wave] = ss; }
    __syncthreads();
    if (tid == 0) {
        float S = 0.f, Q = 0.f;
        #pragma unroll
        for (int i = 0; i < 8; ++i) { S += ws_s[i]; Q += ws_q[i]; }
        float mu  = S * (1.0f / HIDDEN);
        float ex2 = Q * (1.0f / HIDDEN);
        stat[0] = mu;
        stat[1] = __builtin_amdgcn_rsqf(ex2 - mu * mu + EPSLN);
    }
    __syncthreads();
    const float mu = stat[0], rstd = stat[1];

    ushort_t* ho = hout + (size_t)blockIdx.x * PRUNED;
    for (int j = tid; j < PRUNED; j += 256) {
        int idx = s3[j];
        float xv = bf2f(row[idx]);
        float nrm = bf2f(f2bf((xv - mu) * rstd));     // match bf16 cast order
        ho[j] = f2bf(nrm * bf2f(lnw[idx]) + bf2f(lnb[idx]));
    }
}

// ---------------------------------------------------------------------------
// K2: dual GEMM (gate & up share A) + sigmoid/relu epilogue
//   block = 256 threads (8 waves as 2x4), tile 128x128, k-step 32,
//   double-buffered LDS fed by global_load_async_to_lds_b128.
// ---------------------------------------------------------------------------
#define LDSROW 40   // 32 + 8 pad (elements)

__global__ void __launch_bounds__(256)
gemm_gateup_kernel(const ushort_t* __restrict__ h,
                   const ushort_t* __restrict__ gw,
                   const ushort_t* __restrict__ gb,
                   const ushort_t* __restrict__ uw,
                   const ushort_t* __restrict__ ub,
                   ushort_t* __restrict__ act) {
    __shared__ alignas(16) ushort_t As[2][128 * LDSROW];
    __shared__ alignas(16) ushort_t Bg[2][128 * LDSROW];
    __shared__ alignas(16) ushort_t Bu[2][128 * LDSROW];

    const int m0 = blockIdx.y * 128;
    const int n0 = blockIdx.x * 128;
    const int tid  = threadIdx.x;
    const int lane = tid & 31;
    const int wave = tid >> 5;
    const int wm = wave >> 2;          // 0..1  (64 rows each)
    const int wn = wave & 3;           // 0..3  (32 cols each)
    const int lh   = lane >> 4;        // 0/1
    const int ln16 = lane & 15;

    v8f accg[4][2], accu[4][2];
    const v8f vzero = {0.f,0.f,0.f,0.f,0.f,0.f,0.f,0.f};
    #pragma unroll
    for (int i = 0; i < 4; ++i)
        #pragma unroll
        for (int j = 0; j < 2; ++j) { accg[i][j] = vzero; accu[i][j] = vzero; }

    const int r = tid >> 2;            // 0..63
    const int q = (tid & 3) * 8;       // element offset in 32-wide k-slab

    const int nrA = n0 + r;
    const int nrB = n0 + r + 64;
    const int nrAc = nrA < (INTER - 1) ? nrA : (INTER - 1);
    const int nrBc = nrB < (INTER - 1) ? nrB : (INTER - 1);

    // per-thread global row bases (k advances by +k0)
    const ushort_t* hA = h  + (size_t)(m0 + r) * PRUNED + q;
    const ushort_t* hB = hA + (size_t)64 * PRUNED;
    const ushort_t* gA = gw + (size_t)nrAc * PRUNED + q;
    const ushort_t* gB = gw + (size_t)nrBc * PRUNED + q;
    const ushort_t* uA = uw + (size_t)nrAc * PRUNED + q;
    const ushort_t* uB = uw + (size_t)nrBc * PRUNED + q;

    // per-thread LDS destinations (both buffers)
    uint32_t dA0[2], dA1[2], dG0[2], dG1[2], dU0[2], dU1[2];
    #pragma unroll
    for (int b = 0; b < 2; ++b) {
        dA0[b] = lds_off(&As[b][r * LDSROW + q]);
        dA1[b] = lds_off(&As[b][(r + 64) * LDSROW + q]);
        dG0[b] = lds_off(&Bg[b][r * LDSROW + q]);
        dG1[b] = lds_off(&Bg[b][(r + 64) * LDSROW + q]);
        dU0[b] = lds_off(&Bu[b][r * LDSROW + q]);
        dU1[b] = lds_off(&Bu[b][(r + 64) * LDSROW + q]);
    }
    auto issue = [&](int b, int k0) {
        async_b128(dA0[b], hA + k0);
        async_b128(dA1[b], hB + k0);
        async_b128(dG0[b], gA + k0);
        async_b128(dG1[b], gB + k0);
        async_b128(dU0[b], uA + k0);
        async_b128(dU1[b], uB + k0);
    };

    issue(0, 0);
    int buf = 0;
    for (int k0 = 0; k0 < PRUNED; k0 += 32, buf ^= 1) {
        wait_async0();
        __syncthreads();                    // buf ready; prev readers done
        if (k0 + 32 < PRUNED) issue(buf ^ 1, k0 + 32);

        AFrag a[4];
        #pragma unroll
        for (int mf = 0; mf < 4; ++mf) {
            const ushort_t* p = &As[buf][(wm * 64 + mf * 16 + ln16) * LDSROW];
            a[mf].q[0] = *(const uint4*)(p + lh * 16);       // K = lh*16..+7
            a[mf].q[1] = *(const uint4*)(p + lh * 16 + 8);   // K = lh*16+8..+7
        }
        AFrag bgf[2], buf2[2];
        #pragma unroll
        for (int nf = 0; nf < 2; ++nf) {
            const int rowb = (wn * 32 + nf * 16 + ln16) * LDSROW + lh * 16;
            bgf[nf].q[0]  = *(const uint4*)(&Bg[buf][rowb]);
            bgf[nf].q[1]  = *(const uint4*)(&Bg[buf][rowb + 8]);
            buf2[nf].q[0] = *(const uint4*)(&Bu[buf][rowb]);
            buf2[nf].q[1] = *(const uint4*)(&Bu[buf][rowb + 8]);
        }
        #pragma unroll
        for (int mf = 0; mf < 4; ++mf)
            #pragma unroll
            for (int nf = 0; nf < 2; ++nf) {
                accg[mf][nf] = __builtin_amdgcn_wmma_f32_16x16x32_bf16(
                    false, a[mf].v, false, bgf[nf].v, (short)0, accg[mf][nf], false, false);
                accu[mf][nf] = __builtin_amdgcn_wmma_f32_16x16x32_bf16(
                    false, a[mf].v, false, buf2[nf].v, (short)0, accu[mf][nf], false, false);
            }
    }

    // ---- epilogue: act = sigmoid(g+gb) * relu(u+ub); pad cols -> 0 --------
    #pragma unroll
    for (int nf = 0; nf < 2; ++nf) {
        const int n = n0 + wn * 32 + nf * 16 + ln16;
        const int nc = n < (INTER - 1) ? n : (INTER - 1);
        const float gbv = bf2f(gb[nc]);
        const float ubv = bf2f(ub[nc]);
        const bool valid = (n < INTER);
        #pragma unroll
        for (int mf = 0; mf < 4; ++mf) {
            CFrag cg, cu; cg.v = accg[mf][nf]; cu.v = accu[mf][nf];
            const int tbase = m0 + wm * 64 + mf * 16 + lh * 8;
            #pragma unroll
            for (int rr = 0; rr < 8; ++rr) {
                // sigmoid via v_exp + v_rcp (no IEEE divide sequence)
                float e  = __builtin_amdgcn_exp2f(-(cg.f[rr] + gbv) * LOG2E);
                float sg = __builtin_amdgcn_rcpf(1.0f + e);
                float ux = cu.f[rr] + ubv; ux = ux > 0.f ? ux : 0.f;
                float v  = valid ? sg * ux : 0.f;            // branch-free
                act[(size_t)(tbase + rr) * INTER_P + n] = f2bf(v);
            }
        }
    }
}

// ---------------------------------------------------------------------------
// K3: down GEMM + bias + residual add + scatter(s5)
// ---------------------------------------------------------------------------
__global__ void __launch_bounds__(256)
gemm_down_kernel(const ushort_t* __restrict__ act,
                 const ushort_t* __restrict__ dw,
                 const ushort_t* __restrict__ db,
                 const int* __restrict__ s5,
                 const ushort_t* __restrict__ resid,
                 ushort_t* __restrict__ out) {
    __shared__ alignas(16) ushort_t As[2][128 * LDSROW];
    __shared__ alignas(16) ushort_t Bd[2][128 * LDSROW];

    const int m0 = blockIdx.y * 128;
    const int n0 = blockIdx.x * 128;    // pruned-dim tile
    const int tid  = threadIdx.x;
    const int lane = tid & 31;
    const int wave = tid >> 5;
    const int wm = wave >> 2;
    const int wn = wave & 3;
    const int lh   = lane >> 4;
    const int ln16 = lane & 15;

    v8f acc[4][2];
    const v8f vzero = {0.f,0.f,0.f,0.f,0.f,0.f,0.f,0.f};
    #pragma unroll
    for (int i = 0; i < 4; ++i)
        #pragma unroll
        for (int j = 0; j < 2; ++j) acc[i][j] = vzero;

    const int r = tid >> 2;
    const int q = (tid & 3) * 8;

    const ushort_t* aA = act + (size_t)(m0 + r) * INTER_P + q;
    const ushort_t* aB = aA + (size_t)64 * INTER_P;
    const ushort_t* w0 = dw + (size_t)(n0 + r) * INTER;        // rows 8B-aligned
    const ushort_t* w1 = dw + (size_t)(n0 + r + 64) * INTER;

    uint32_t dA0[2], dA1[2], dB0[2], dB1[2];
    #pragma unroll
    for (int b = 0; b < 2; ++b) {
        dA0[b] = lds_off(&As[b][r * LDSROW + q]);
        dA1[b] = lds_off(&As[b][(r + 64) * LDSROW + q]);
        dB0[b] = lds_off(&Bd[b][r * LDSROW + q]);
        dB1[b] = lds_off(&Bd[b][(r + 64) * LDSROW + q]);
    }
    auto issue = [&](int b, int k0) {
        async_b128(dA0[b], aA + k0);
        async_b128(dA1[b], aB + k0);
        // clamp k so reads stay inside the INTER-long rows; clamped lanes
        // at ks>=5008 multiply zero-padded act cols -> 0; at ks=5000 act cols 5000..5003 get w[4996..4999]
        int ks = k0 + q;
        int ksc = ks < (INTER - 8) ? ks : (INTER - 8);
        async_b64(dB0[b],     w0 + ksc);
        async_b64(dB0[b] + 8, w0 + ksc + 4);
        async_b64(dB1[b],     w1 + ksc);
        async_b64(dB1[b] + 8, w1 + ksc + 4);
    };

    issue(0, 0);
    int buf = 0;
    for (int k0 = 0; k0 < INTER_P; k0 += 32, buf ^= 1) {
        wait_async0();
        __syncthreads();
        if (k0 + 32 < INTER_P) issue(buf ^ 1, k0 + 32);

        AFrag a[4];
        #pragma unroll
        for (int mf = 0; mf < 4; ++mf) {
            const ushort_t* p = &As[buf][(wm * 64 + mf * 16 + ln16) * LDSROW];
            a[mf].q[0] = *(const uint4*)(p + lh * 16);
            a[mf].q[1] = *(const uint4*)(p + lh * 16 + 8);
        }
        AFrag bf[2];
        #pragma unroll
        for (int nf = 0; nf < 2; ++nf) {
            const int rowb = (wn * 32 + nf * 16 + ln16) * LDSROW + lh * 16;
            bf[nf].q[0] = *(const uint4*)(&Bd[buf][rowb]);
            bf[nf].q[1] = *(const uint4*)(&Bd[buf][rowb + 8]);
        }
        #pragma unroll
        for (int mf = 0; mf < 4; ++mf)
            #pragma unroll
            for (int nf = 0; nf < 2; ++nf)
                acc[mf][nf] = __builtin_amdgcn_wmma_f32_16x16x32_bf16(
                    false, a[mf].v, false, bf[nf].v, (short)0, acc[mf][nf], false, false);
    }

    // ---- epilogue: out[t, s5[j]] = resid + mlp + down_b ----
    #pragma unroll
    for (int nf = 0; nf < 2; ++nf) {
        const int j = n0 + wn * 32 + nf * 16 + ln16;      // 0..2047
        const int col = s5[j];
        const float dbv = bf2f(db[j]);
        #pragma unroll
        for (int mf = 0; mf < 4; ++mf) {
            CFrag c; c.v = acc[mf][nf];
            const int tbase = m0 + wm * 64 + mf * 16 + lh * 8;
            #pragma unroll
            for (int rr = 0; rr < 8; ++rr) {
                const size_t o = (size_t)(tbase + rr) * HIDDEN + col;
                out[o] = f2bf(bf2f(resid[o]) + c.f[rr] + dbv);
            }
        }
    }
}

// ---------------------------------------------------------------------------
// Host entry
// ---------------------------------------------------------------------------
extern "C" void kernel_launch(void* const* d_in, const int* in_sizes, int n_in,
                              void* d_out, int out_size, void* d_ws, size_t ws_size,
                              hipStream_t stream) {
    const ushort_t* hidden = (const ushort_t*)d_in[0];
    const ushort_t* ln_w   = (const ushort_t*)d_in[1];
    const ushort_t* ln_b   = (const ushort_t*)d_in[2];
    const ushort_t* gate_w = (const ushort_t*)d_in[3];
    const ushort_t* gate_b = (const ushort_t*)d_in[4];
    const ushort_t* up_w   = (const ushort_t*)d_in[5];
    const ushort_t* up_b   = (const ushort_t*)d_in[6];
    const ushort_t* down_w = (const ushort_t*)d_in[7];
    const ushort_t* down_b = (const ushort_t*)d_in[8];
    const int*      s3     = (const int*)d_in[9];
    const int*      s5     = (const int*)d_in[10];
    ushort_t*       out    = (ushort_t*)d_out;

    // workspace: h [MTOK x PRUNED] (64 MiB) then act [MTOK x INTER_P] (160 MiB)
    ushort_t* h_ws   = (ushort_t*)d_ws;
    ushort_t* act_ws = h_ws + (size_t)MTOK * PRUNED;

    {
        const size_t nvec = (size_t)MTOK * HIDDEN / 8;   // uint4 count
        copy_residual_kernel<<<dim3((unsigned)(nvec / 256)), dim3(256), 0, stream>>>(
            (const uint4*)hidden, (uint4*)out);
    }
    ln_gather_kernel<<<dim3(MTOK), dim3(256), 0, stream>>>(
        hidden, ln_w, ln_b, s3, h_ws);
    gemm_gateup_kernel<<<dim3(INTER_P / 128, MTOK / 128), dim3(256), 0, stream>>>(
        h_ws, gate_w, gate_b, up_w, up_b, act_ws);
    gemm_down_kernel<<<dim3(PRUNED / 128, MTOK / 128), dim3(256), 0, stream>>>(
        act_ws, down_w, down_b, s5, hidden, out);
}
